// CausalMultiHeadSelfAttention_25563645346145
// MI455X (gfx1250) — matrix operand order repaired and checked by reading, not compiled, not run on hardware
//
#include <hip/hip_runtime.h>
#include <hip/hip_bf16.h>

// CDNA5 / gfx1250 wave32 WMMA implementation of causal MHA.
// All matmuls via v_wmma_f32_16x16x32_bf16 (fp32 accumulate).
// Global->LDS staging uses GLOBAL_LOAD_ASYNC_TO_LDS_B128 (ASYNCcnt-tracked),
// software-pipelined with double-buffered LDS.

typedef __bf16 bf16_t;
typedef __attribute__((ext_vector_type(16))) __bf16 v16bf;
typedef __attribute__((ext_vector_type(8)))  __bf16 v8bf;
typedef __attribute__((ext_vector_type(8)))  float  v8f;
typedef __attribute__((ext_vector_type(4)))  int    v4i;

#define D_MODEL   1024
#define NUM_HEADS 16
#define HEAD_DIM  64
#define BATCH     4
#define SEQ       2048

#define GAS __attribute__((address_space(1)))
#define LAS __attribute__((address_space(3)))

#if defined(__gfx1250__) && __has_builtin(__builtin_amdgcn_global_load_async_to_lds_b128)
#define HAS_ASYNC_LDS 1
#else
#define HAS_ASYNC_LDS 0
#endif

// 16-byte global -> LDS copy (async on CDNA5; ASYNCcnt-tracked).
__device__ __forceinline__ void g2l_b128(const bf16_t* g, bf16_t* l) {
#if HAS_ASYNC_LDS
  __builtin_amdgcn_global_load_async_to_lds_b128(
      (GAS v4i*)(unsigned long long)g,
      (LAS v4i*)(unsigned int)(unsigned long long)l, 0, 0);
#else
  *(v8bf*)l = *(const v8bf*)g;
#endif
}

__device__ __forceinline__ void wait_async() {
#if HAS_ASYNC_LDS
#if __has_builtin(__builtin_amdgcn_s_wait_asynccnt)
  __builtin_amdgcn_s_wait_asynccnt(0);
#else
  asm volatile("s_wait_asynccnt 0x0" ::: "memory");
#endif
#endif
}

__device__ __forceinline__ v8f zero_v8f() {
  v8f z;
#pragma unroll
  for (int i = 0; i < 8; ++i) z[i] = 0.0f;
  return z;
}

__device__ __forceinline__ v16bf cat8(v8bf lo, v8bf hi) {
  return __builtin_shufflevector(lo, hi, 0, 1, 2, 3, 4, 5, 6, 7,
                                 8, 9, 10, 11, 12, 13, 14, 15);
}

// ---------------------------------------------------------------------------
// fp32 -> bf16 convert (grid-stride)
// ---------------------------------------------------------------------------
__global__ void f32_to_bf16_kernel(const float* __restrict__ src,
                                   bf16_t* __restrict__ dst, size_t n) {
  for (size_t i = (size_t)blockIdx.x * blockDim.x + threadIdx.x; i < n;
       i += (size_t)gridDim.x * blockDim.x) {
    dst[i] = (bf16_t)src[i];
  }
}

// ---------------------------------------------------------------------------
// NT GEMM: C[M,N] = A[M,K] * B[N,K]^T   (A,B bf16 row-major, K contiguous)
// Workgroup: 128 threads = 4 waves (2x2). WG tile 128x128; wave tile 64x64.
// K staged 32 at a time via async-to-LDS, double-buffered so the async DMA of
// tile k+1 overlaps the 16 WMMA of tile k. One barrier per K-step.
// ---------------------------------------------------------------------------
template <bool BF16_OUT>
__global__ __launch_bounds__(128) void gemm_nt_bf16(
    const bf16_t* __restrict__ A, const bf16_t* __restrict__ Bm,
    void* __restrict__ Cv, int M, int N, int K) {
  __shared__ __align__(32) bf16_t lA[2][128 * 32];
  __shared__ __align__(32) bf16_t lB[2][128 * 32];

  const int tid  = threadIdx.x;
  const int wave = tid >> 5;
  const int lane = tid & 31;
  const int nl   = lane & 15;   // row/col-in-tile index
  const int hh   = lane >> 4;   // half-wave select

  const int m0 = blockIdx.x * 128;
  const int n0 = blockIdx.y * 128;
  const int mw = 64 * (wave >> 1);  // wave sub-tile origin in M
  const int nw = 64 * (wave & 1);   // wave sub-tile origin in N

  auto stage = [&](int k0, int buf) {
    const bf16_t* ag = A + (size_t)(m0 + tid) * K + k0;
    const bf16_t* bg = Bm + (size_t)(n0 + tid) * K + k0;
    bf16_t* la = &lA[buf][tid * 32];
    bf16_t* lb = &lB[buf][tid * 32];
#pragma unroll
    for (int c = 0; c < 4; ++c) {
      g2l_b128(ag + 8 * c, la + 8 * c);
      g2l_b128(bg + 8 * c, lb + 8 * c);
    }
  };

  v8f acc[4][4];
#pragma unroll
  for (int i = 0; i < 4; ++i)
#pragma unroll
    for (int j = 0; j < 4; ++j) acc[i][j] = zero_v8f();

  stage(0, 0);
  wait_async();
  __syncthreads();

  for (int k0 = 0; k0 < K; k0 += 32) {
    const int cur = (k0 >> 5) & 1;
    // issue async DMA for next K-step while computing on the current one
    if (k0 + 32 < K) stage(k0 + 32, cur ^ 1);
    if (k0 + 64 < K) {  // cache prefetch two K-steps ahead
      __builtin_prefetch(A + (size_t)(m0 + tid) * K + k0 + 64, 0, 1);
      __builtin_prefetch(Bm + (size_t)(n0 + tid) * K + k0 + 64, 0, 1);
    }

    // ---- A fragments: ISA 16-bit A layout (k = 8h..8h+7, 16+8h..16+8h+7) ----
    v16bf afrag[4];
#pragma unroll
    for (int ci = 0; ci < 4; ++ci) {
      const bf16_t* ar = &lA[cur][(mw + 16 * ci + nl) * 32];
      v8bf lo = *(const v8bf*)(ar + 8 * hh);
      v8bf hi = *(const v8bf*)(ar + 16 + 8 * hh);
      afrag[ci] = cat8(lo, hi);
    }
    // ---- B fragments: lane holds column n, same K order as A ----
#pragma unroll
    for (int cj = 0; cj < 4; ++cj) {
      const bf16_t* br = &lB[cur][(nw + 16 * cj + nl) * 32];
      v16bf bfrag = cat8(*(const v8bf*)(br + 8 * hh),
                         *(const v8bf*)(br + 16 + 8 * hh));
#pragma unroll
      for (int ci = 0; ci < 4; ++ci) {
        acc[ci][cj] = __builtin_amdgcn_wmma_f32_16x16x32_bf16(
            false, afrag[ci], false, bfrag, (short)0, acc[ci][cj], false,
            false);
      }
    }
    wait_async();
    __syncthreads();
  }

  // ---- store: C tile row m = r + 8*hh, col n = nl ----
#pragma unroll
  for (int ci = 0; ci < 4; ++ci) {
#pragma unroll
    for (int cj = 0; cj < 4; ++cj) {
      const int gn = n0 + nw + 16 * cj + nl;
      const int gm = m0 + mw + 16 * ci + 8 * hh;
#pragma unroll
      for (int r = 0; r < 8; ++r) {
        if (BF16_OUT)
          ((bf16_t*)Cv)[(size_t)(gm + r) * N + gn] = (bf16_t)acc[ci][cj][r];
        else
          ((float*)Cv)[(size_t)(gm + r) * N + gn] = acc[ci][cj][r];
      }
    }
  }
}

// ---------------------------------------------------------------------------
// RoPE (q,k) + head-major transpose; folds 1/sqrt(dh) into q.
// [B,S,D] -> [B,H,S,dh]
// ---------------------------------------------------------------------------
__global__ void rope_transpose_kernel(
    const bf16_t* __restrict__ qlin, const bf16_t* __restrict__ klin,
    const bf16_t* __restrict__ vlin, const int* __restrict__ pos,
    bf16_t* __restrict__ qh, bf16_t* __restrict__ kh,
    bf16_t* __restrict__ vh) {
  const int total = BATCH * SEQ * NUM_HEADS * (HEAD_DIM / 2);
  for (int idx = blockIdx.x * blockDim.x + threadIdx.x; idx < total;
       idx += gridDim.x * blockDim.x) {
    const int i = idx & 31;           // rotary pair index
    const int h = (idx >> 5) & 15;    // head
    const int s = (idx >> 9) & 2047;  // sequence position
    const int b = idx >> 20;          // batch

    const size_t lin = (size_t)(b * SEQ + s) * D_MODEL + h * HEAD_DIM + 2 * i;
    const size_t hd =
        ((size_t)(b * NUM_HEADS + h) * SEQ + s) * HEAD_DIM + 2 * i;

    const float p = (float)pos[s];
    // inv_freq = theta^(-2i/dh); ln(10000) = 9.2103403...
    const float freq =
        __expf(-((float)(2 * i) / (float)HEAD_DIM) * 9.210340371976184f);
    float sn, cs;
    __sincosf(p * freq, &sn, &cs);

    const float qs = 0.125f;  // 1/sqrt(64)
    const float q1 = (float)qlin[lin], q2 = (float)qlin[lin + 1];
    qh[hd]     = (bf16_t)((q1 * cs - q2 * sn) * qs);
    qh[hd + 1] = (bf16_t)((q1 * sn + q2 * cs) * qs);
    const float k1 = (float)klin[lin], k2 = (float)klin[lin + 1];
    kh[hd]     = (bf16_t)(k1 * cs - k2 * sn);
    kh[hd + 1] = (bf16_t)(k1 * sn + k2 * cs);
    vh[hd]     = vlin[lin];
    vh[hd + 1] = vlin[lin + 1];
  }
}

// ---------------------------------------------------------------------------
// Causal flash attention. Grid: (S/64, H, B), 128 threads = 4 waves.
// Each wave owns 16 query rows. Computes S^T = K*Q^T so each query is a
// C-tile COLUMN: row stats need only per-lane reduction + shfl_xor(16).
// O accumulated transposed: O^T = V^T * P^T (same per-lane query mapping,
// so online-softmax rescale is a per-lane scalar multiply).
// K/V staging double-buffered; K via async-to-LDS.
// ---------------------------------------------------------------------------
__global__ __launch_bounds__(128) void flash_attn_kernel(
    const bf16_t* __restrict__ qh, const bf16_t* __restrict__ kh,
    const bf16_t* __restrict__ vh, bf16_t* __restrict__ attn /*[B,S,H,dh]*/) {
  __shared__ __align__(32) bf16_t lK[2][32 * 64];   // [k][d]
  __shared__ __align__(32) bf16_t lVt[2][64 * 32];  // [d][k] (transposed)
  __shared__ __align__(32) bf16_t lP[4][16 * 32];   // per-wave P^T as [q][k]

  const int tid = threadIdx.x;
  const int w = tid >> 5, lane = tid & 31, nl = lane & 15, hh = lane >> 4;
  const int qb = blockIdx.x * 64;
  const int h  = blockIdx.y;
  const int b  = blockIdx.z;

  const size_t head = (size_t)(b * NUM_HEADS + h) * SEQ * HEAD_DIM;
  const bf16_t* Qp = qh + head;
  const bf16_t* Kp = kh + head;
  const bf16_t* Vp = vh + head;

  const int q0   = qb + 16 * w;
  const int qcol = q0 + nl;  // this lane's query row (tile column)

  const int kr = tid & 31, half = tid >> 5;
  auto stage = [&](int k0, int buf) {
    const bf16_t* kg = Kp + (size_t)(k0 + kr) * HEAD_DIM + 16 * half;
    g2l_b128(kg, &lK[buf][kr * 64 + 16 * half]);
    g2l_b128(kg + 8, &lK[buf][kr * 64 + 16 * half + 8]);
    const int ds = 16 * half;
    v16bf vv = *(const v16bf*)(Vp + (size_t)(k0 + kr) * HEAD_DIM + ds);
#pragma unroll
    for (int i = 0; i < 16; ++i) lVt[buf][(ds + i) * 32 + kr] = vv[i];
  };

  // Q kept in registers all loop: B-operand frags for the two 32-wide K-chunks
  v16bf bq[2];
#pragma unroll
  for (int c = 0; c < 2; ++c)
    bq[c] = cat8(*(const v8bf*)(Qp + (size_t)(q0 + nl) * HEAD_DIM + 32 * c +
                                8 * hh),
                 *(const v8bf*)(Qp + (size_t)(q0 + nl) * HEAD_DIM + 32 * c +
                                16 + 8 * hh));

  v8f o[4];
#pragma unroll
  for (int t = 0; t < 4; ++t) o[t] = zero_v8f();
  float mrow = -1e30f, lrow = 0.0f;

  const int nblk = (qb + 64) / 32;  // causal: only key blocks <= q block end
  stage(0, 0);
  wait_async();
  __syncthreads();

  for (int jb = 0; jb < nblk; ++jb) {
    const int k0  = jb * 32;
    const int cur = jb & 1;
    // issue staging of the next key block while computing on this one
    if (jb + 1 < nblk) stage(k0 + 32, cur ^ 1);

    // ---- S^T tiles: two 16(k) x 16(q) tiles, contraction over dh=64 ----
    v8f st[2];
#pragma unroll
    for (int ti = 0; ti < 2; ++ti) {
      st[ti] = zero_v8f();
      const bf16_t* krow = &lK[cur][(16 * ti + nl) * 64];
#pragma unroll
      for (int c = 0; c < 2; ++c) {
        v8bf lo = *(const v8bf*)(krow + 32 * c + 8 * hh);
        v8bf hi = *(const v8bf*)(krow + 32 * c + 16 + 8 * hh);
        st[ti] = __builtin_amdgcn_wmma_f32_16x16x32_bf16(
            false, cat8(lo, hi), false, bq[c], (short)0, st[ti], false, false);
      }
    }
    // causal mask only when this key block can cross the wave's diagonal
    // (wave-uniform branch; at most the last 2 blocks of the loop)
    if (k0 + 31 > q0) {
#pragma unroll
      for (int ti = 0; ti < 2; ++ti)
#pragma unroll
        for (int r = 0; r < 8; ++r) {
          const int kidx = k0 + 16 * ti + 8 * hh + r;
          if (kidx > qcol) st[ti][r] = -1e30f;
        }
    }

    // ---- online softmax per query column (lane pair l <-> l^16) ----
    float tmax = -1e30f;
#pragma unroll
    for (int ti = 0; ti < 2; ++ti)
#pragma unroll
      for (int r = 0; r < 8; ++r) tmax = fmaxf(tmax, st[ti][r]);
    tmax = fmaxf(tmax, __shfl_xor(tmax, 16, 32));
    const float mnew  = fmaxf(mrow, tmax);
    const float alpha = __expf(mrow - mnew);

    float tsum = 0.0f;
#pragma unroll
    for (int ti = 0; ti < 2; ++ti) {
      v8bf pv;
#pragma unroll
      for (int r = 0; r < 8; ++r) {
        const float e = __expf(st[ti][r] - mnew);
        tsum += e;
        pv[r] = (bf16_t)e;
      }
      // C-layout rows are contiguous per lane: one 16B LDS store per tile
      *(v8bf*)&lP[w][nl * 32 + 16 * ti + 8 * hh] = pv;
    }
    tsum += __shfl_xor(tsum, 16, 32);
    lrow = lrow * alpha + tsum;
    mrow = mnew;
#pragma unroll
    for (int t = 0; t < 4; ++t) o[t] *= alpha;

    // ---- O^T += V^T(16d x 32k) * P^T(32k x 16q), per d-tile ----
    v16bf pf = cat8(*(const v8bf*)(&lP[w][nl * 32 + 8 * hh]),
                    *(const v8bf*)(&lP[w][nl * 32 + 16 + 8 * hh]));
#pragma unroll
    for (int t = 0; t < 4; ++t) {
      const bf16_t* vr = &lVt[cur][(16 * t + nl) * 32];
      v8bf lo = *(const v8bf*)(vr + 8 * hh);
      v8bf hi = *(const v8bf*)(vr + 16 + 8 * hh);
      o[t] = __builtin_amdgcn_wmma_f32_16x16x32_bf16(
          false, cat8(lo, hi), false, pf, (short)0, o[t], false, false);
    }
    wait_async();
    __syncthreads();
  }

  // ---- normalize and write [B,S,H,dh] bf16 (contiguous 8-element runs) ----
  const float inv = 1.0f / lrow;
#pragma unroll
  for (int t = 0; t < 4; ++t) {
    v8bf ov;
#pragma unroll
    for (int r = 0; r < 8; ++r) ov[r] = (bf16_t)(o[t][r] * inv);
    bf16_t* dst = attn +
                  ((size_t)(b * SEQ + qcol) * NUM_HEADS + h) * HEAD_DIM +
                  16 * t + 8 * hh;
    *(v8bf*)dst = ov;
  }
}

// ---------------------------------------------------------------------------
// Host-side orchestration
// ---------------------------------------------------------------------------
extern "C" void kernel_launch(void* const* d_in, const int* in_sizes, int n_in,
                              void* d_out, int out_size, void* d_ws,
                              size_t ws_size, hipStream_t stream) {
  (void)in_sizes; (void)n_in; (void)out_size; (void)ws_size;
  const float* x  = (const float*)d_in[0];
  const float* Wq = (const float*)d_in[1];
  const float* Wk = (const float*)d_in[2];
  const float* Wv = (const float*)d_in[3];
  const float* Wo = (const float*)d_in[4];
  const int* pos  = (const int*)d_in[5];

  const int M = BATCH * SEQ;  // 8192
  const int N = D_MODEL, K = D_MODEL;
  const size_t AEL = (size_t)M * D_MODEL;        // activation elements
  const size_t WEL = (size_t)D_MODEL * D_MODEL;  // weight elements

  char* wsp = (char*)d_ws;
  size_t off = 0;
  auto take = [&](size_t elems) {
    bf16_t* p = (bf16_t*)(wsp + off);
    off += ((elems * sizeof(bf16_t)) + 255) & ~(size_t)255;
    return p;
  };
  bf16_t* xb    = take(AEL);
  bf16_t* Wqb   = take(WEL);
  bf16_t* Wkb   = take(WEL);
  bf16_t* Wvb   = take(WEL);
  bf16_t* Wob   = take(WEL);
  bf16_t* qlin  = take(AEL);
  bf16_t* klin  = take(AEL);
  bf16_t* vlin  = take(AEL);
  bf16_t* qhb   = take(AEL);
  bf16_t* khb   = take(AEL);
  bf16_t* vhb   = take(AEL);
  bf16_t* attnb = take(AEL);

  // 1. fp32 -> bf16
  f32_to_bf16_kernel<<<2048, 256, 0, stream>>>(x, xb, AEL);
  f32_to_bf16_kernel<<<512, 256, 0, stream>>>(Wq, Wqb, WEL);
  f32_to_bf16_kernel<<<512, 256, 0, stream>>>(Wk, Wkb, WEL);
  f32_to_bf16_kernel<<<512, 256, 0, stream>>>(Wv, Wvb, WEL);
  f32_to_bf16_kernel<<<512, 256, 0, stream>>>(Wo, Wob, WEL);

  // 2. Q/K/V projections (x * W^T), bf16 out
  dim3 gg(M / 128, N / 128);
  gemm_nt_bf16<true><<<gg, 128, 0, stream>>>(xb, Wqb, qlin, M, N, K);
  gemm_nt_bf16<true><<<gg, 128, 0, stream>>>(xb, Wkb, klin, M, N, K);
  gemm_nt_bf16<true><<<gg, 128, 0, stream>>>(xb, Wvb, vlin, M, N, K);

  // 3. RoPE + transpose to [B,H,S,dh]
  rope_transpose_kernel<<<4096, 256, 0, stream>>>(qlin, klin, vlin, pos, qhb,
                                                  khb, vhb);

  // 4. causal flash attention -> [B,S,H,dh] bf16
  dim3 ga(SEQ / 64, NUM_HEADS, BATCH);
  flash_attn_kernel<<<ga, 128, 0, stream>>>(qhb, khb, vhb, attnb);

  // 5. output projection (attn * Wo^T), fp32 out
  gemm_nt_bf16<false><<<gg, 128, 0, stream>>>(attnb, Wob, (float*)d_out, M, N,
                                              K);
}
